// DeformableConvolutionV3_79714593014176
// MI455X (gfx1250) — matrix operand order repaired and checked by reading, not compiled, not run on hardware
//
#include <hip/hip_runtime.h>
#include <hip/hip_bf16.h>
#include <math.h>

// ---------------------------------------------------------------------------
// DeformableConvV3 for MI455X (gfx1250, wave32, WMMA)
// B=8 H=64 W=64 C=256 G=8 GC=32 K=3 K2=9 PAD=1
// All GEMMs: f16 operands, fp32 accumulate on v_wmma_f32_16x16x32_f16.
// Packed weights are zero-padded to a multiple of 4 N-tiles so the GEMM main
// loop is branch-free (EXEC all-ones, accumulators pinned in VGPRs).
// ---------------------------------------------------------------------------

typedef __attribute__((ext_vector_type(16))) _Float16 v16h;
typedef __attribute__((ext_vector_type(8)))  _Float16 v8h;
typedef __attribute__((ext_vector_type(4)))  _Float16 v4h;
typedef __attribute__((ext_vector_type(8)))  float    v8f;
typedef __attribute__((ext_vector_type(4)))  float    v4f;

#define BATCH 8
#define HH    64
#define WW    64
#define CC    256
#define GG    8
#define GC_   32
#define K2_   9
#define PAD_  1
#define HP    66
#define WP    66
#define MM    (BATCH * HH * WW)   // 32768 "pixels" (GEMM M)
#define KD    256                 // GEMM K (= C)
#define KSTEPS 8                  // K / 32

// ---------------------------------------------------------------------------
// Zero-fill (for padded x_proj ring)
// ---------------------------------------------------------------------------
__global__ void zero_f32(float* __restrict__ p, long long n) {
    long long i = (long long)blockIdx.x * blockDim.x + threadIdx.x;
    long long stride = (long long)gridDim.x * blockDim.x;
    for (; i < n; i += stride) p[i] = 0.0f;
}

// ---------------------------------------------------------------------------
// One-shot fp32 -> f16 conversion (4 elements / thread / iter)
// ---------------------------------------------------------------------------
__global__ void f32_to_f16(const float* __restrict__ src, _Float16* __restrict__ dst,
                           long long n4) {
    long long i = (long long)blockIdx.x * blockDim.x + threadIdx.x;
    long long stride = (long long)gridDim.x * blockDim.x;
    for (; i < n4; i += stride) {
        v4f v = ((const v4f*)src)[i];
        v4h h;
        #pragma unroll
        for (int j = 0; j < 4; ++j) h[j] = (_Float16)v[j];
        ((v4h*)dst)[i] = h;
    }
}

// ---------------------------------------------------------------------------
// Pack fp32 weight matrix (K x Nreal, row-major) into f16 WMMA B fragments.
// Fragment layout per 16x16x32 step: lane l holds N = l&15, halves i=0..15
// cover K = 16*(l>>4) + i within the 32-wide K step. Stored contiguously:
//   dst[(((ntile*KSTEPS)+ks)*32 + lane)*16 + i]
// Ntiles is rounded up to a multiple of 4; columns >= Nreal are zero.
// ---------------------------------------------------------------------------
__global__ void pack_weights(const float* __restrict__ Wsrc, _Float16* __restrict__ dst,
                             int Nreal, int Ntiles) {
    int tid = blockIdx.x * blockDim.x + threadIdx.x;
    int total = Ntiles * KSTEPS * 32 * 16;
    if (tid >= total) return;
    int i     =  tid        & 15;
    int lane  = (tid >> 4)  & 31;
    int ks    = (tid >> 9)  & 7;
    int ntile =  tid >> 12;
    int k = ks * 32 + ((lane >> 4) * 16) + i;
    int n = ntile * 16 + (lane & 15);
    float v = (n < Nreal) ? Wsrc[(size_t)k * Nreal + n] : 0.0f;
    dst[tid] = (_Float16)v;
}

// ---------------------------------------------------------------------------
// WMMA GEMM: C[M x Nreal] = A[M x 256] (f16) * Bpacked + bias
// One wave -> a 16(M) x 64(N) strip (4 N-tiles, always): each A fragment
// feeds 4 v_wmma_f32_16x16x32_f16 (4x A reuse). 4 waves/block cover 4
// consecutive M-tiles, sharing B fragments via L0/L2. The main loop is
// branch-free; only epilogue stores are guarded (col < Nreal).
//   - PADOUT: scatter rows into the zero-padded (B,66,66,256) x_proj buffer
// ---------------------------------------------------------------------------
template <bool PADOUT>
__global__ void __launch_bounds__(128)
gemm_wmma(const _Float16* __restrict__ A, const _Float16* __restrict__ Bp,
          const float* __restrict__ bias, float* __restrict__ Cout,
          int Nreal, int ostride) {
    const int lane  = threadIdx.x & 31;
    const int wave  = threadIdx.x >> 5;
    const int mtile = blockIdx.x * 4 + wave;
    const int nt0   = blockIdx.y * 4;
    const int m0    = mtile * 16;
    const int khalf = (lane >> 4) * 16;    // A: lanes 16-31 take the K+16 halves
    const int mrow  = m0 + (lane & 15);

    const _Float16* arow = A + (size_t)mrow * KD;
    // Per-lane base of the 4 consecutive N-tile fragment streams.
    const _Float16* bl = Bp + ((size_t)nt0 * KSTEPS * 32 + lane) * 16;

    v8f acc0 = {}, acc1 = {}, acc2 = {}, acc3 = {};

    #pragma unroll
    for (int ks = 0; ks < KSTEPS; ++ks) {
        const _Float16* ap = arow + ks * 32 + khalf;
        v8h lo = *(const v8h*)(ap);
        v8h hi = *(const v8h*)(ap + 8);
        if (ks + 1 < KSTEPS) __builtin_prefetch(ap + 32, 0, 3);
        v16h a;
        #pragma unroll
        for (int j = 0; j < 8; ++j) { a[j] = lo[j]; a[8 + j] = hi[j]; }

        v16h b0 = *(const v16h*)(bl + (size_t)(0 * KSTEPS + ks) * 512);
        v16h b1 = *(const v16h*)(bl + (size_t)(1 * KSTEPS + ks) * 512);
        v16h b2 = *(const v16h*)(bl + (size_t)(2 * KSTEPS + ks) * 512);
        v16h b3 = *(const v16h*)(bl + (size_t)(3 * KSTEPS + ks) * 512);

        acc0 = __builtin_amdgcn_wmma_f32_16x16x32_f16(false, a, false, b0,
                                                      (short)0, acc0, false, false);
        acc1 = __builtin_amdgcn_wmma_f32_16x16x32_f16(false, a, false, b1,
                                                      (short)0, acc1, false, false);
        acc2 = __builtin_amdgcn_wmma_f32_16x16x32_f16(false, a, false, b2,
                                                      (short)0, acc2, false, false);
        acc3 = __builtin_amdgcn_wmma_f32_16x16x32_f16(false, a, false, b3,
                                                      (short)0, acc3, false, false);
    }

    // Epilogue. D layout: lanes 0-15: VGPR r -> M=r; lanes 16-31: M=8+r; N=lane&15.
    const int n     = lane & 15;
    const int mbase = (lane >> 4) * 8;
    #pragma unroll
    for (int t = 0; t < 4; ++t) {
        const v8f& acc = (t == 0) ? acc0 : (t == 1) ? acc1 : (t == 2) ? acc2 : acc3;
        int col = (nt0 + t) * 16 + n;
        if (col >= Nreal) continue;   // phantom zero-padded columns
        float bv = bias[col];
        #pragma unroll
        for (int r = 0; r < 8; ++r) {
            int gm  = m0 + mbase + r;
            float v = acc[r] + bv;
            if constexpr (PADOUT) {
                int bb = gm >> 12;
                int hh = (gm >> 6) & 63;
                int ww = gm & 63;
                Cout[(((size_t)bb * HP + (hh + 1)) * WP + (ww + 1)) * CC + col] = v;
            } else {
                Cout[(size_t)gm * ostride + col] = v;
            }
        }
    }
}

// ---------------------------------------------------------------------------
// Depthwise 3x3 (SAME, zero pad) + LayerNorm(C) + exact GELU -> x1 (f16)
// One block per pixel, 256 threads = channels, LDS tree reductions.
// ---------------------------------------------------------------------------
__global__ void __launch_bounds__(256)
dw_ln_gelu(const float* __restrict__ x, const float* __restrict__ dwk,
           const float* __restrict__ dwb, const float* __restrict__ gamma,
           const float* __restrict__ beta, _Float16* __restrict__ x1h) {
    __shared__ float red[256];
    const int pix = blockIdx.x;
    const int c   = threadIdx.x;
    const int b   =  pix >> 12;
    const int h   = (pix >> 6) & 63;
    const int w   =  pix & 63;

    float s = dwb[c];
    #pragma unroll
    for (int ky = 0; ky < 3; ++ky) {
        int hh = h + ky - 1;
        if (hh < 0 || hh >= HH) continue;
        #pragma unroll
        for (int kx = 0; kx < 3; ++kx) {
            int ww = w + kx - 1;
            if (ww < 0 || ww >= WW) continue;
            s += x[(((size_t)b * HH + hh) * WW + ww) * CC + c] *
                 dwk[(ky * 3 + kx) * CC + c];
        }
    }

    red[c] = s;
    __syncthreads();
    for (int off = 128; off > 0; off >>= 1) {
        if (c < off) red[c] += red[c + off];
        __syncthreads();
    }
    float mu = red[0] * (1.0f / 256.0f);
    __syncthreads();
    float d = s - mu;
    red[c] = d * d;
    __syncthreads();
    for (int off = 128; off > 0; off >>= 1) {
        if (c < off) red[c] += red[c + off];
        __syncthreads();
    }
    float var = red[0] * (1.0f / 256.0f);

    float y = d * rsqrtf(var + 1e-6f) * gamma[c] + beta[c];
    float g = 0.5f * y * (1.0f + erff(y * 0.70710678118654752f));
    x1h[(size_t)pix * CC + c] = (_Float16)g;
}

// ---------------------------------------------------------------------------
// Softmax over 9 taps + deformable bilinear gather, mask-weighted sum.
// One wave per (pixel, group); 32 lanes = 32 group channels. Output f16 for
// the final WMMA GEMM. x_pad has the zero SAME-padding ring baked in; taps
// landing outside the padded frame are masked per-corner (matches the
// reference's valid*weight semantics).
// ---------------------------------------------------------------------------
__global__ void __launch_bounds__(256)
deform_gather(const float* __restrict__ xpad, const float* __restrict__ off,
              const float* __restrict__ msk, _Float16* __restrict__ outh) {
    const int lane = threadIdx.x & 31;
    const int wid  = blockIdx.x * (blockDim.x >> 5) + (threadIdx.x >> 5);
    const int g    = wid & 7;
    const int pix  = wid >> 3;
    const int b    =  pix >> 12;
    const int h    = (pix >> 6) & 63;
    const int w    =  pix & 63;

    const float* op = off + (size_t)pix * 144 + g * 18;  // (k,2) pairs, x then y
    const float* mp = msk + (size_t)pix * 80  + g * 9;   // logits (stride-80 rows)

    float e[9];
    float mx = -3.4e38f;
    #pragma unroll
    for (int k = 0; k < 9; ++k) { e[k] = mp[k]; mx = fmaxf(mx, e[k]); }
    float sum = 0.0f;
    #pragma unroll
    for (int k = 0; k < 9; ++k) { e[k] = __expf(e[k] - mx); sum += e[k]; }
    const float inv = 1.0f / sum;

    const float* xg = xpad + (size_t)b * HP * WP * CC + g * GC_ + lane;
    float acc = 0.0f;
    #pragma unroll
    for (int k = 0; k < 9; ++k) {
        float px = (float)(PAD_ + w) + ((float)(k % 3 - 1) + op[2 * k + 0]);
        float py = (float)(PAD_ + h) + ((float)(k / 3 - 1) + op[2 * k + 1]);
        float x0f = floorf(px), y0f = floorf(py);
        float wx = px - x0f, wy = py - y0f;
        int x0 = (int)x0f, y0 = (int)y0f;
        float sv = 0.0f;
        #pragma unroll
        for (int cy = 0; cy < 2; ++cy) {
            int yy = y0 + cy;
            if (yy < 0 || yy >= HP) continue;
            float wyv = cy ? wy : (1.0f - wy);
            #pragma unroll
            for (int cx = 0; cx < 2; ++cx) {
                int xx = x0 + cx;
                if (xx < 0 || xx >= WP) continue;
                float wxv = cx ? wx : (1.0f - wx);
                sv += wyv * wxv * xg[((size_t)yy * WP + xx) * CC];
            }
        }
        acc += sv * e[k] * inv;
    }
    outh[(size_t)pix * CC + g * GC_ + lane] = (_Float16)acc;
}

// ---------------------------------------------------------------------------
// Host launcher
// ---------------------------------------------------------------------------
extern "C" void kernel_launch(void* const* d_in, const int* in_sizes, int n_in,
                              void* d_out, int out_size, void* d_ws, size_t ws_size,
                              hipStream_t stream) {
    const float* inputs = (const float*)d_in[0];
    const float* w_in   = (const float*)d_in[1];
    const float* b_in   = (const float*)d_in[2];
    const float* dw_k   = (const float*)d_in[3];
    const float* dw_b   = (const float*)d_in[4];
    const float* ln_g   = (const float*)d_in[5];
    const float* ln_b   = (const float*)d_in[6];
    const float* w_off  = (const float*)d_in[7];
    const float* b_off  = (const float*)d_in[8];
    const float* w_mask = (const float*)d_in[9];
    const float* b_mask = (const float*)d_in[10];
    const float* w_out  = (const float*)d_in[11];
    const float* b_out  = (const float*)d_in[12];

    char* ws = (char*)d_ws;
    size_t o = 0;
    auto alloc = [&](size_t bytes) -> char* {
        char* p = ws + o;
        o = (o + bytes + 255) & ~(size_t)255;
        return p;
    };
    float*    x_pad = (float*)   alloc((size_t)BATCH * HP * WP * CC * 4);  // 35.7 MB
    _Float16* inh   = (_Float16*)alloc((size_t)MM * CC * 2);               // 16 MB
    _Float16* x1h   = (_Float16*)alloc((size_t)MM * CC * 2);               // 16 MB
    float*    offb  = (float*)   alloc((size_t)MM * 144 * 4);              // 18.9 MB
    float*    mskb  = (float*)   alloc((size_t)MM * 80 * 4);               // 10.5 MB
    _Float16* outh  = (_Float16*)alloc((size_t)MM * CC * 2);               // 16 MB
    _Float16* winP  = (_Float16*)alloc((size_t)16 * 4096 * 2);   // 16 tiles
    _Float16* woffP = (_Float16*)alloc((size_t)12 * 4096 * 2);   // 9 -> 12 tiles
    _Float16* wmskP = (_Float16*)alloc((size_t)8  * 4096 * 2);   // 5 -> 8 tiles
    _Float16* woutP = (_Float16*)alloc((size_t)16 * 4096 * 2);   // 16 tiles

    // 1) zero padded x_proj ring; convert inputs to f16 once
    zero_f32<<<2048, 256, 0, stream>>>(x_pad, (long long)BATCH * HP * WP * CC);
    f32_to_f16<<<2048, 256, 0, stream>>>(inputs, inh, (long long)MM * CC / 4);

    // 2) pack all weights into WMMA B-fragment layout (f16), N-tiles padded to 4
    pack_weights<<<(16 * 4096 + 255) / 256, 256, 0, stream>>>(w_in,   winP, 256, 16);
    pack_weights<<<(12 * 4096 + 255) / 256, 256, 0, stream>>>(w_off,  woffP, 144, 12);
    pack_weights<<<(8  * 4096 + 255) / 256, 256, 0, stream>>>(w_mask, wmskP, 72,  8);
    pack_weights<<<(16 * 4096 + 255) / 256, 256, 0, stream>>>(w_out,  woutP, 256, 16);

    const dim3 blk(128);
    // 3) x_proj = inputs @ w_in + b_in  -> padded buffer   (16 tiles -> y=4)
    gemm_wmma<true><<<dim3(MM / 64, 4), blk, 0, stream>>>(
        inh, winP, b_in, x_pad, 256, 0);

    // 4) depthwise + LN + GELU -> x1 (f16)
    dw_ln_gelu<<<MM, 256, 0, stream>>>(inputs, dw_k, dw_b, ln_g, ln_b, x1h);

    // 5) offsets (12 tiles -> y=3) and mask logits (8 tiles -> y=2)
    gemm_wmma<false><<<dim3(MM / 64, 3), blk, 0, stream>>>(
        x1h, woffP, b_off, offb, 144, 144);
    gemm_wmma<false><<<dim3(MM / 64, 2), blk, 0, stream>>>(
        x1h, wmskP, b_mask, mskb, 72, 80);

    // 6) softmax + deformable bilinear gather -> f16
    deform_gather<<<MM * GG / 8, 256, 0, stream>>>(x_pad, offb, mskb, outh);

    // 7) final projection -> d_out (fp32)
    gemm_wmma<false><<<dim3(MM / 64, 4), blk, 0, stream>>>(
        outh, woutP, b_out, (float*)d_out, 256, 256);
}
